// GatingGCN_18743237280393
// MI455X (gfx1250) — compile-verified, matrix operand order checked by reading, not run on hardware
//
#include <hip/hip_runtime.h>
#include <hip/hip_bf16.h>
#include <math.h>

typedef __attribute__((ext_vector_type(16))) _Float16 v16h;
typedef __attribute__((ext_vector_type(8)))  float    v8f;

#define TPB 256

// ---------------- utility fills ----------------
__global__ void fill_f32(float* __restrict__ p, int n, float v) {
    int i = blockIdx.x * blockDim.x + threadIdx.x;
    if (i < n) p[i] = v;
}

// ---------------- degree / rsqrt ----------------
__global__ void count_deg(const int* __restrict__ dst, float* __restrict__ deg, int E) {
    int i = blockIdx.x * blockDim.x + threadIdx.x;
    if (i < E) atomicAdd(&deg[dst[i]], 1.0f);
}

__global__ void rsqrt_inplace(float* __restrict__ p, int n) {
    int i = blockIdx.x * blockDim.x + threadIdx.x;
    if (i < n) p[i] = rsqrtf(p[i]);
}

// ---------------- WMMA GEMM: Y[rows x 64] = X[rows x 64] @ W[64 x 64] ----------------
// One wave -> one 16x16 tile of Y. rows must be a multiple of 16 (100000 = 6250*16).
// Grid sized so every wave has a valid tile => no divergence, EXEC all-1s for WMMA.
__global__ void gemm64_wmma(const float* __restrict__ X, const float* __restrict__ W,
                            float* __restrict__ Y) {
    const int wave = threadIdx.x >> 5;
    const int lane = threadIdx.x & 31;
    const int tile = blockIdx.x * 8 + wave;   // 8 waves per block
    const int rt   = tile >> 2;               // row tile (16 rows)
    const int ct   = tile & 3;                // col tile (16 cols of 64)
    const int row0 = rt << 4;
    const int col0 = ct << 4;

    const int m    = lane & 15;               // A row / B,D column within tile
    const int half = lane >> 4;
    const int kb   = half * 8;                // K sub-offset per ISA A/B layout

    v8f c = {};
    const float* __restrict__ xrow = X + (row0 + m) * 64;

#pragma unroll
    for (int k0 = 0; k0 < 64; k0 += 32) {
        v16h a, b;
#pragma unroll
        for (int j = 0; j < 8; ++j) {
            a[j]     = (_Float16)xrow[k0 + kb + j];
            a[8 + j] = (_Float16)xrow[k0 + kb + 16 + j];
        }
#pragma unroll
        for (int j = 0; j < 8; ++j) {
            b[j]     = (_Float16)W[(k0 + kb + j)      * 64 + col0 + m];
            b[8 + j] = (_Float16)W[(k0 + kb + 16 + j) * 64 + col0 + m];
        }
        c = __builtin_amdgcn_wmma_f32_16x16x32_f16(false, a, false, b,
                                                   (short)0, c, false, false);
    }

    // D layout: VGPR r -> row (r + 8*half), lane%16 -> column
#pragma unroll
    for (int r = 0; r < 8; ++r) {
        Y[(row0 + r + 8 * half) * 64 + col0 + m] = c[r];
    }
}

// ---------------- edge scatter: agg[dst] += h[src] * dinv[src]*dinv[dst] ----------------
// One thread per (edge, channel); consecutive lanes = consecutive channels of the
// same edge => coalesced loads and contiguous global_atomic_add_f32.
__global__ void edge_agg(const float* __restrict__ h, const int* __restrict__ src,
                         const int* __restrict__ dst, const float* __restrict__ dinv,
                         float* __restrict__ agg, int total /* = E*64 */) {
    int idx = blockIdx.x * blockDim.x + threadIdx.x;
    if (idx >= total) return;
    int e = idx >> 6;
    int c = idx & 63;
    int s = src[e];
    int d = dst[e];
    float w = dinv[s] * dinv[d];
    atomicAdd(&agg[d * 64 + c], h[s * 64 + c] * w);
}

// ---------------- finalize: out = [relu](agg + h*dinv^2 + bias) ----------------
__global__ void finalize_layer(const float* __restrict__ agg, const float* __restrict__ h,
                               const float* __restrict__ dinv, const float* __restrict__ bias,
                               float* __restrict__ out, int total /* = N*64 */, int do_relu) {
    int idx = blockIdx.x * blockDim.x + threadIdx.x;
    if (idx >= total) return;
    int i = idx >> 6;
    int c = idx & 63;
    float di = dinv[i];
    float v = agg[idx] + h[idx] * di * di + bias[c];
    out[idx] = do_relu ? fmaxf(v, 0.0f) : v;
}

// ---------------- head + segment pooling ----------------
// Per node: logits = h2[i,:] @ Wl + bl (64x8), atomically accumulated into its graph.
__global__ void logits_pool(const float* __restrict__ h2, const float* __restrict__ Wl,
                            const float* __restrict__ bl, const int* __restrict__ batch,
                            float* __restrict__ pooled, float* __restrict__ cnt, int n) {
    int i = blockIdx.x * blockDim.x + threadIdx.x;
    if (i >= n) return;
    float acc[8];
#pragma unroll
    for (int k = 0; k < 8; ++k) acc[k] = bl[k];
    const float* __restrict__ row = h2 + i * 64;
#pragma unroll 4
    for (int c = 0; c < 64; ++c) {
        float x = row[c];
#pragma unroll
        for (int k = 0; k < 8; ++k) acc[k] += x * Wl[c * 8 + k];
    }
    int g = batch[i];
#pragma unroll
    for (int k = 0; k < 8; ++k) atomicAdd(&pooled[g * 8 + k], acc[k]);
    atomicAdd(&cnt[g], 1.0f);
}

// ---------------- mean + log_softmax over K=8, one thread per graph ----------------
__global__ void mean_log_softmax(const float* __restrict__ pooled, const float* __restrict__ cnt,
                                 float* __restrict__ out, int G) {
    int g = blockIdx.x * blockDim.x + threadIdx.x;
    if (g >= G) return;
    float inv = 1.0f / fmaxf(cnt[g], 1.0f);
    float v[8];
    float mx = -3.402823466e+38f;
#pragma unroll
    for (int k = 0; k < 8; ++k) {
        v[k] = pooled[g * 8 + k] * inv;
        mx = fmaxf(mx, v[k]);
    }
    float s = 0.0f;
#pragma unroll
    for (int k = 0; k < 8; ++k) s += __expf(v[k] - mx);
    float ls = __logf(s);
#pragma unroll
    for (int k = 0; k < 8; ++k) out[g * 8 + k] = v[k] - mx - ls;
}

extern "C" void kernel_launch(void* const* d_in, const int* in_sizes, int n_in,
                              void* d_out, int out_size, void* d_ws, size_t ws_size,
                              hipStream_t stream) {
    const float* x     = (const float*)d_in[0];
    const int*   ei    = (const int*)  d_in[1];
    const int*   batch = (const int*)  d_in[2];
    const float* W1    = (const float*)d_in[3];
    const float* b1    = (const float*)d_in[4];
    const float* W2    = (const float*)d_in[5];
    const float* b2    = (const float*)d_in[6];
    const float* Wl    = (const float*)d_in[7];
    const float* bl    = (const float*)d_in[8];
    float* out = (float*)d_out;

    const int N = in_sizes[0] / 64;   // 100000
    const int E = in_sizes[1] / 2;    // 1250000
    const int G = out_size / 8;       // 256
    const int* src = ei;
    const int* dst = ei + E;

    // workspace partition (floats)
    float* ws     = (float*)d_ws;
    float* dinv   = ws;                         // N
    float* h      = dinv + N;                   // N*64  (GEMM output, per layer)
    float* agg    = h    + (size_t)N * 64;      // N*64
    float* hout   = agg  + (size_t)N * 64;      // N*64  (finalized layer output)
    float* pooled = hout + (size_t)N * 64;      // G*8
    float* cnt    = pooled + (size_t)G * 8;     // G

    const int NH      = N * 64;                 // 6.4M
    const int EH      = E * 64;                 // 80M
    const int nTiles  = (N / 16) * 4;           // 25000
    const int gemmBlk = nTiles / 8;             // 3125 blocks * 8 waves

    auto cdiv = [](int a, int b) { return (a + b - 1) / b; };

    // --- degree / dinv ---
    fill_f32<<<cdiv(N, TPB), TPB, 0, stream>>>(dinv, N, 1.0f);      // self loop
    count_deg<<<cdiv(E, TPB), TPB, 0, stream>>>(dst, dinv, E);
    rsqrt_inplace<<<cdiv(N, TPB), TPB, 0, stream>>>(dinv, N);

    // --- layer 1: h = x @ W1 ; agg ; relu(agg + h*dinv^2 + b1) ---
    gemm64_wmma<<<gemmBlk, TPB, 0, stream>>>(x, W1, h);
    fill_f32<<<cdiv(NH, TPB), TPB, 0, stream>>>(agg, NH, 0.0f);
    edge_agg<<<cdiv(EH, TPB), TPB, 0, stream>>>(h, src, dst, dinv, agg, EH);
    finalize_layer<<<cdiv(NH, TPB), TPB, 0, stream>>>(agg, h, dinv, b1, hout, NH, 1);

    // --- layer 2: h = hout @ W2 ; agg ; agg + h*dinv^2 + b2 ---
    gemm64_wmma<<<gemmBlk, TPB, 0, stream>>>(hout, W2, h);
    fill_f32<<<cdiv(NH, TPB), TPB, 0, stream>>>(agg, NH, 0.0f);
    edge_agg<<<cdiv(EH, TPB), TPB, 0, stream>>>(h, src, dst, dinv, agg, EH);
    finalize_layer<<<cdiv(NH, TPB), TPB, 0, stream>>>(agg, h, dinv, b2, hout, NH, 0);

    // --- head + pooling + log_softmax ---
    fill_f32<<<cdiv(G * 9, TPB), TPB, 0, stream>>>(pooled, G * 9, 0.0f); // pooled + cnt contiguous
    logits_pool<<<cdiv(N, TPB), TPB, 0, stream>>>(hout, Wl, bl, batch, pooled, cnt, N);
    mean_log_softmax<<<cdiv(G, TPB), TPB, 0, stream>>>(pooled, cnt, out, G);
}
